// MaskedMultiHeadAttention_90091234001081
// MI455X (gfx1250) — matrix operand order repaired and checked by reading, not compiled, not run on hardware
//
#include <hip/hip_runtime.h>
#include <stdint.h>

typedef __bf16 bf16_t;
typedef __attribute__((ext_vector_type(16))) __bf16 v16bf;
typedef __attribute__((ext_vector_type(8)))  __bf16 v8bf;
typedef __attribute__((ext_vector_type(8)))  float  v8f;

constexpr int Bn = 2, Ssz = 2048, Esz = 1024, Hn = 16, Dd = 64;
constexpr int Msz = Bn * Ssz;   // 4096 rows total
constexpr int Ksz = Esz;        // 1024
constexpr int Nsz = Esz;        // 1024

constexpr int LPITCH = 40;            // LDS row pitch in bf16 elems (80 B, 16B-aligned)
constexpr int ABUF = 128 * LPITCH;    // elems per A staging buffer (128 rows x 32 k)
constexpr int BBUF = 64 * LPITCH;     // elems per B staging buffer (64 rows x 32 k)

// ---------- helpers ----------
__device__ __forceinline__ v8bf ld8(const bf16_t* p) {
  return *reinterpret_cast<const v8bf*>(p);
}
__device__ __forceinline__ v16bf mk16(v8bf lo, v8bf hi) {
  union { v8bf h[2]; v16bf v; } u;
  u.h[0] = lo; u.h[1] = hi;
  return u.v;
}
__device__ __forceinline__ v8f wmma_bf16(v16bf a, v16bf b, v8f c) {
  // D = A(16x32 bf16) * B(32x16 bf16) + C(16x16 f32)
  return __builtin_amdgcn_wmma_f32_16x16x32_bf16(false, a, false, b,
                                                 (short)0, c, false, false);
}
// gfx1250 async DMA: 16B global -> LDS, tracked by ASYNCcnt (ISA 15.18.3 op 98)
__device__ __forceinline__ void async_ld_b128(uint32_t lds_off, const bf16_t* g) {
  asm volatile("global_load_async_to_lds_b128 %0, %1, off"
               :: "v"(lds_off), "v"((unsigned long long)(uintptr_t)g)
               : "memory");
}
__device__ __forceinline__ void wait_async0() {
  asm volatile("s_wait_asynccnt 0x0" ::: "memory");
}
__device__ __forceinline__ float hmax16(float v) {
  v = fmaxf(v, __shfl_xor(v, 1, 32));
  v = fmaxf(v, __shfl_xor(v, 2, 32));
  v = fmaxf(v, __shfl_xor(v, 4, 32));
  v = fmaxf(v, __shfl_xor(v, 8, 32));
  return v;
}
__device__ __forceinline__ float hsum16(float v) {
  v += __shfl_xor(v, 1, 32);
  v += __shfl_xor(v, 2, 32);
  v += __shfl_xor(v, 4, 32);
  v += __shfl_xor(v, 8, 32);
  return v;
}

// ---------- fp32 -> bf16 conversions ----------
__global__ void cvt_bf16_kernel(const float* __restrict__ in,
                                bf16_t* __restrict__ out, int n) {
  int i = blockIdx.x * blockDim.x + threadIdx.x;
  if (i < n) out[i] = (bf16_t)in[i];
}

// in[rows][cols] f32 -> out[cols][rows] bf16 (weight pre-transpose)
__global__ void cvtT_bf16_kernel(const float* __restrict__ in,
                                 bf16_t* __restrict__ out, int rows, int cols) {
  int i = blockIdx.x * blockDim.x + threadIdx.x;
  if (i < rows * cols) {
    int r = i / cols, c = i % cols;
    out[(size_t)c * rows + r] = (bf16_t)in[i];
  }
}

// ---------- WMMA GEMM: C[M,N] = A[M,K](bf16 rowmajor) * Bt[N,K](bf16) ----------
// Block = 128(M) x 64(N), 8 waves, 32x32 per wave. K-loop step 32, double-buffered
// async-DMA staging of A/B tiles into LDS (global_load_async_to_lds_b128).
// mode 0: store bf16 row-major [M,N]
// mode 1: store bf16 per-head-transposed Vt[(b*Hn+h)*Dd+d][s]
// mode 2: store f32 row-major + bias
__global__ __launch_bounds__(256)
void gemm_wmma_kernel(const bf16_t* __restrict__ A, const bf16_t* __restrict__ Bt,
                      bf16_t* __restrict__ outb, float* __restrict__ outf,
                      const float* __restrict__ bias, int mode) {
  __shared__ __align__(16) bf16_t As[2 * ABUF];
  __shared__ __align__(16) bf16_t Bs[2 * BBUF];

  const int tid = threadIdx.x;
  const int lane = tid & 31;
  const int wave = tid >> 5;
  const int Hh = lane >> 4;      // half-wave select
  const int ln = lane & 15;
  const int mblk = blockIdx.y * 128;
  const int nblk = blockIdx.x * 64;
  const int wm = (wave & 3) * 32;
  const int wn = (wave >> 2) * 32;

  // flat->LDS: low 32 bits of a generic pointer to __shared__ are the LDS offset
  const uint32_t asBase = (uint32_t)(uintptr_t)(void*)As;
  const uint32_t bsBase = (uint32_t)(uintptr_t)(void*)Bs;

  // per-thread staging slices:
  //   A tile 128x32: thread t -> row t>>1, 32 contiguous bytes at (t&1)*32
  //   B tile  64x32: thread t -> row t>>2, 16 contiguous bytes at (t&3)*16
  const size_t gaRow = (size_t)(mblk + (tid >> 1)) * Ksz + (size_t)((tid & 1) * 16);
  const size_t gbRow = (size_t)(nblk + (tid >> 2)) * Ksz + (size_t)((tid & 3) * 8);
  const uint32_t laOff = (uint32_t)((tid >> 1) * (LPITCH * 2) + (tid & 1) * 32);
  const uint32_t lbOff = (uint32_t)((tid >> 2) * (LPITCH * 2) + (tid & 3) * 16);

  auto issue = [&](int k0, int buf) {
    const bf16_t* ga = A + gaRow + k0;
    const uint32_t la = asBase + (uint32_t)buf * (ABUF * 2) + laOff;
    async_ld_b128(la, ga);
    async_ld_b128(la + 16, ga + 8);
    const bf16_t* gb = Bt + gbRow + k0;
    const uint32_t lb = bsBase + (uint32_t)buf * (BBUF * 2) + lbOff;
    async_ld_b128(lb, gb);
  };

  v8f acc[2][2] = {};
  const int nsteps = Ksz / 32;

  issue(0, 0);
  for (int st = 0; st < nsteps; ++st) {
    wait_async0();       // my loads for buf[st&1] have landed in LDS
    __syncthreads();     // ...and so have everyone else's
    if (st + 1 < nsteps) issue((st + 1) * 32, (st + 1) & 1);  // overlap with math

    const bf16_t* Ab = As + (st & 1) * ABUF;
    const bf16_t* Bb = Bs + (st & 1) * BBUF;
    v16bf a[2], b[2];
#pragma unroll
    for (int i = 0; i < 2; ++i) {
      // A fragment: lane=row, 16 consecutive K at 16H
      const bf16_t* pa = Ab + (wm + 16 * i + ln) * LPITCH + 16 * Hh;
      a[i] = mk16(ld8(pa), ld8(pa + 8));
      // B fragment: lane=col, 16 consecutive K at 16H
      const bf16_t* pb = Bb + (wn + 16 * i + ln) * LPITCH + 16 * Hh;
      b[i] = mk16(ld8(pb), ld8(pb + 8));
    }
#pragma unroll
    for (int i = 0; i < 2; ++i)
#pragma unroll
      for (int j = 0; j < 2; ++j)
        acc[i][j] = wmma_bf16(a[i], b[j], acc[i][j]);
  }

#pragma unroll
  for (int i = 0; i < 2; ++i) {
    const int mb = mblk + wm + 16 * i + 8 * Hh;   // first row this lane owns
#pragma unroll
    for (int j = 0; j < 2; ++j) {
      const int col = nblk + wn + 16 * j + ln;
      if (mode == 0) {
#pragma unroll
        for (int r = 0; r < 8; ++r)
          outb[(size_t)(mb + r) * Nsz + col] = (bf16_t)acc[i][j][r];
      } else if (mode == 1) {
        const int h = col >> 6, d = col & 63;
        const int bb = mb >> 11, s = mb & (Ssz - 1);
        v8bf pk;
#pragma unroll
        for (int r = 0; r < 8; ++r) pk[r] = (bf16_t)acc[i][j][r];
        // rows m -> consecutive s: one contiguous 16B store
        *reinterpret_cast<v8bf*>(outb + ((size_t)((bb * Hn + h) * Dd + d)) * Ssz + s) = pk;
      } else {
        const float bv = bias[col];
#pragma unroll
        for (int r = 0; r < 8; ++r)
          outf[(size_t)(mb + r) * Nsz + col] = acc[i][j][r] + bv;
      }
    }
  }
}

// ---------- flash attention: one wave per (b, h, 16-query tile) ----------
__global__ __launch_bounds__(128)
void attn_wmma_kernel(const bf16_t* __restrict__ Q, const bf16_t* __restrict__ Kc,
                      const bf16_t* __restrict__ Vt, bf16_t* __restrict__ O) {
  __shared__ __align__(16) bf16_t Pl[4][16 * 40];  // per-wave P buffer, 80B row pitch
  const int lane = threadIdx.x & 31, wave = threadIdx.x >> 5;
  const int Hh = lane >> 4, ln = lane & 15;
  const int gw = blockIdx.x * 4 + wave;
  const int qtiles = Ssz / 16;                 // 128
  const int bh = gw / qtiles, qt = gw % qtiles;
  const int q0 = qt * 16;
  const int b = bh / Hn, h = bh % Hn;

  const bf16_t* Qh = Q + (size_t)b * Ssz * Esz + h * Dd;   // row stride Esz
  const bf16_t* Kh = Kc + (size_t)b * Ssz * Esz + h * Dd;  // row stride Esz
  const bf16_t* Vh = Vt + (size_t)bh * Dd * Ssz;           // Dd rows, stride Ssz
  bf16_t* Pw = Pl[wave];

  // Q tile as two A fragments (d chunks 0..31, 32..63)
  v16bf aq[2];
#pragma unroll
  for (int c = 0; c < 2; ++c) {
    const bf16_t* pq = Qh + (size_t)(q0 + ln) * Esz + 32 * c + 16 * Hh;
    aq[c] = mk16(ld8(pq), ld8(pq + 8));
  }

  v8f Oacc[4] = {};
  float rmax[8], rsum[8];
#pragma unroll
  for (int r = 0; r < 8; ++r) { rmax[r] = -3.0e38f; rsum[r] = 0.0f; }

  const int nk = ((q0 + 15) >> 5) + 1;  // causal: 32-key steps covering keys <= q0+15
  for (int kt = 0; kt < nk; ++kt) {
    const int k0 = kt * 32;
    // scores: two 16x16 tiles (keys k0..+15, k0+16..+31), 2 WMMAs each over d
    v8f st[2] = {};
#pragma unroll
    for (int t = 0; t < 2; ++t) {
      const bf16_t* pk = Kh + (size_t)(k0 + 16 * t + ln) * Esz + 16 * Hh;
      v16bf bk0 = mk16(ld8(pk), ld8(pk + 8));            // d 0..31
      v16bf bk1 = mk16(ld8(pk + 32), ld8(pk + 40));      // d 32..63
      st[t] = wmma_bf16(aq[0], bk0, st[t]);
      st[t] = wmma_bf16(aq[1], bk1, st[t]);
    }
    // scale 1/sqrt(64), causal mask, online softmax over this 32-key block
#pragma unroll
    for (int r = 0; r < 8; ++r) {
      const int qrow = q0 + 8 * Hh + r;
      float s0 = st[0][r] * 0.125f; if (k0 + ln > qrow)      s0 = -3.0e38f;
      float s1 = st[1][r] * 0.125f; if (k0 + 16 + ln > qrow) s1 = -3.0e38f;
      const float mx = hmax16(fmaxf(s0, s1));
      const float nm = fmaxf(rmax[r], mx);
      const float al = __expf(rmax[r] - nm);
      const float p0 = __expf(s0 - nm);
      const float p1 = __expf(s1 - nm);
      rsum[r] = rsum[r] * al + hsum16(p0 + p1);
      rmax[r] = nm;
#pragma unroll
      for (int c = 0; c < 4; ++c) Oacc[c][r] *= al;
      Pw[(8 * Hh + r) * 40 + ln]      = (bf16_t)p0;
      Pw[(8 * Hh + r) * 40 + 16 + ln] = (bf16_t)p1;
    }
    // re-stripe P (C-layout f32 -> A-layout bf16) via LDS; same-wave ordering
    asm volatile("s_wait_dscnt 0x0" ::: "memory");
    v16bf pf = mk16(*reinterpret_cast<const v8bf*>(Pw + ln * 40 + 16 * Hh),
                    *reinterpret_cast<const v8bf*>(Pw + ln * 40 + 8 + 16 * Hh));
    // O += P(16x32) * V(32x64): 4 WMMAs, B rows contiguous thanks to Vt layout
#pragma unroll
    for (int c = 0; c < 4; ++c) {
      const bf16_t* pv = Vh + (size_t)(16 * c + ln) * Ssz + k0 + 16 * Hh;
      v16bf bv = mk16(ld8(pv), ld8(pv + 8));
      Oacc[c] = wmma_bf16(pf, bv, Oacc[c]);
    }
  }

  float inv[8];
#pragma unroll
  for (int r = 0; r < 8; ++r) inv[r] = 1.0f / rsum[r];
  bf16_t* Ob = O + (size_t)b * Ssz * Esz + h * Dd;
#pragma unroll
  for (int r = 0; r < 8; ++r) {
    const int srow = q0 + 8 * Hh + r;
#pragma unroll
    for (int c = 0; c < 4; ++c)
      Ob[(size_t)srow * Esz + 16 * c + ln] = (bf16_t)(Oacc[c][r] * inv[r]);
  }
}

// ---------- launch ----------
extern "C" void kernel_launch(void* const* d_in, const int* in_sizes, int n_in,
                              void* d_out, int out_size, void* d_ws, size_t ws_size,
                              hipStream_t stream) {
  const float* x  = (const float*)d_in[0];
  const float* Wq = (const float*)d_in[1];
  const float* Wk = (const float*)d_in[2];
  const float* Wv = (const float*)d_in[3];
  const float* Wo = (const float*)d_in[4];
  const float* bo = (const float*)d_in[5];
  float* out = (float*)d_out;

  char* ws = (char*)d_ws;
  auto take = [&](size_t bytes) -> char* {
    char* p = ws;
    ws += (bytes + 255) & ~size_t(255);
    return p;
  };
  bf16_t* xb  = (bf16_t*)take((size_t)Msz * Esz * 2);  // 8 MB
  bf16_t* WqT = (bf16_t*)take((size_t)Esz * Esz * 2);  // 2 MB each
  bf16_t* WkT = (bf16_t*)take((size_t)Esz * Esz * 2);
  bf16_t* WvT = (bf16_t*)take((size_t)Esz * Esz * 2);
  bf16_t* WoT = (bf16_t*)take((size_t)Esz * Esz * 2);
  bf16_t* Qb  = (bf16_t*)take((size_t)Msz * Esz * 2);  // [B,S,E]
  bf16_t* Kb  = (bf16_t*)take((size_t)Msz * Esz * 2);  // [B,S,E]
  bf16_t* Vtb = (bf16_t*)take((size_t)Msz * Esz * 2);  // [B*H*D, S]
  bf16_t* Oat = (bf16_t*)take((size_t)Msz * Esz * 2);  // [B,S,E]
  (void)ws_size; (void)in_sizes; (void)n_in; (void)out_size;

  const int nx = Msz * Esz;        // 4,194,304
  cvt_bf16_kernel<<<nx / 256, 256, 0, stream>>>(x, xb, nx);
  const int nw = Esz * Esz;        // 1,048,576
  cvtT_bf16_kernel<<<nw / 256, 256, 0, stream>>>(Wq, WqT, Esz, Esz);
  cvtT_bf16_kernel<<<nw / 256, 256, 0, stream>>>(Wk, WkT, Esz, Esz);
  cvtT_bf16_kernel<<<nw / 256, 256, 0, stream>>>(Wv, WvT, Esz, Esz);
  cvtT_bf16_kernel<<<nw / 256, 256, 0, stream>>>(Wo, WoT, Esz, Esz);

  dim3 gg(Nsz / 64, Msz / 128);    // (16, 32)
  gemm_wmma_kernel<<<gg, 256, 0, stream>>>(xb, WqT, Qb,  nullptr, nullptr, 0);
  gemm_wmma_kernel<<<gg, 256, 0, stream>>>(xb, WkT, Kb,  nullptr, nullptr, 0);
  gemm_wmma_kernel<<<gg, 256, 0, stream>>>(xb, WvT, Vtb, nullptr, nullptr, 1);

  attn_wmma_kernel<<<(Bn * Hn * (Ssz / 16)) / 4, 128, 0, stream>>>(Qb, Kb, Vtb, Oat);

  gemm_wmma_kernel<<<gg, 256, 0, stream>>>(Oat, WoT, nullptr, out, bo, 2);
}
